// PartialRadialLayer_77807627534804
// MI455X (gfx1250) — matrix operand order repaired and checked by reading, not compiled, not run on hardware
//
#include <hip/hip_runtime.h>
#include <math.h>

typedef __attribute__((ext_vector_type(16))) _Float16 v16h;
typedef __attribute__((ext_vector_type(8)))  _Float16 v8h;
typedef __attribute__((ext_vector_type(8)))  float    v8f;

#define B_SZ 2048
#define D_SZ 512
#define W_SZ 512
#define L_SZ 64          // bins
#define NN   63          // inner nodes
#define K_SZ (L_SZ * D_SZ)   // 32768

// ---------------------------------------------------------------------------
// Kernel 1: soft bin distribution  dist[B, 64]
// ---------------------------------------------------------------------------
__global__ __launch_bounds__(256) void prl_dist_kernel(
    const float* __restrict__ x, const float* __restrict__ ray,
    const float* __restrict__ w_i, const float* __restrict__ b_i,
    const float* __restrict__ a_i, const int* __restrict__ idx,
    float* __restrict__ dist)
{
    __shared__ float st[8][2 * NN];
    const int lane = threadIdx.x & 31;
    const int wave = threadIdx.x >> 5;
    const int b    = blockIdx.x * 8 + wave;

    const float* xr = x + (size_t)b * D_SZ;
    float dot = 0.f, nx = 0.f, nr = 0.f;
    for (int i = lane; i < D_SZ; i += 32) {
        float xv = xr[i], rv = ray[i];
        dot = fmaf(xv, rv, dot);
        nx  = fmaf(xv, xv, nx);
        nr  = fmaf(rv, rv, nr);
    }
    #pragma unroll
    for (int off = 16; off > 0; off >>= 1) {
        dot += __shfl_xor(dot, off, 32);
        nx  += __shfl_xor(nx,  off, 32);
        nr  += __shfl_xor(nr,  off, 32);
    }
    const float eps = 1e-8f;
    float cosv = dot / (fmaxf(sqrtf(nx), eps) * fmaxf(sqrtf(nr), eps));
    cosv = fminf(1.f, fmaxf(-1.f, cosv));
    const float angle = acosf(cosv) * 0.3183098861837907f;

    for (int j = lane; j < NN; j += 32) {
        float sw = 1.f / (1.f + expf(-w_i[j]));
        float sb = 1.f / (1.f + expf(-b_i[j]));
        float nf = (0.5f + sw) * angle - sb;
        float d  = 1.f / (1.f + expf(-nf * (1.f + a_i[j])));
        st[wave][j]      = d;
        st[wave][j + NN] = 1.f - d;
    }
    __syncthreads();
    for (int leaf = lane; leaf < L_SZ; leaf += 32) {
        float p = 1.f;
        #pragma unroll
        for (int d = 0; d < 6; d++) p *= st[wave][idx[leaf * 6 + d]];
        dist[(size_t)b * L_SZ + leaf] = p;
    }
}

// ---------------------------------------------------------------------------
// Kernel 2: convert + transpose weights: Tt[n][k] (f16) = T[k][n] (f32)
// ---------------------------------------------------------------------------
__global__ __launch_bounds__(256) void prl_cvt_kernel(
    const float* __restrict__ T, _Float16* __restrict__ Tt)
{
    __shared__ _Float16 tile[64][64 + 8];
    const int k0  = blockIdx.x * 64;
    const int n0  = blockIdx.y * 64;
    const int c   = threadIdx.x & 63;
    const int r0  = threadIdx.x >> 6;     // 0..3
    #pragma unroll
    for (int r = r0; r < 64; r += 4)
        tile[r][c] = (_Float16)T[(size_t)(k0 + r) * W_SZ + n0 + c];
    __syncthreads();
    #pragma unroll
    for (int n = r0; n < 64; n += 4)
        Tt[(size_t)(n0 + n) * K_SZ + k0 + c] = tile[c][n];
}

// ---------------------------------------------------------------------------
// Kernel 3: out[2048,512] = A[2048,32768] @ T[32768,512]  via f16 WMMA
//   A[b, l*512+i] = dist[b,l] * x[b,i]  (fp32 scale, then cvt to f16 in LDS)
// BM=BN=BK=64, 128 threads = 4 waves, each wave 32x32 (4 accumulators).
// LDS double-buffered; staging is fused (load->cvt->store) so no registers
// stay live across the WMMA block -> no scratch spills.
// ---------------------------------------------------------------------------
#define BM 64
#define BN 64
#define BK 64
#define LDP (BK + 8)              // padded row: 72 halves = 144 B
#define NSTAGE (K_SZ / BK)        // 512

__global__ __launch_bounds__(128) void prl_gemm_f16_kernel(
    const float* __restrict__ x, const float* __restrict__ dist,
    const _Float16* __restrict__ Tt, float* __restrict__ out)
{
    __shared__ _Float16 As[2][BM][LDP];   // [buf][m][k]
    __shared__ _Float16 Bs[2][BN][LDP];   // [buf][n][k]  (K-major: contiguous frags)

    const int bm   = blockIdx.x * BM;
    const int bn   = blockIdx.y * BN;
    const int tid  = threadIdx.x;
    const int lane = tid & 31;
    const int wave = tid >> 5;           // 0..3
    const int wm   = (wave & 1) * 32;
    const int wn   = (wave >> 1) * 32;

    // WMMA 16-bit fragment addressing (ISA VGPR layout tables)
    const int fm  = lane & 15;
    const int hi  = lane >> 4;           // 0/1
    const int akb = hi * 8;              // A: halves [akb..+7] and [akb+16..+7]
    const int bkb = hi * 8;              // B: halves [bkb..+7] and [bkb+16..+7]

    v8f c00 = {}, c01 = {}, c10 = {}, c11 = {};

    // cooperative staging map (identical for A and B):
    const int ar = tid >> 1;             // 0..63  (row m for A, row n for B)
    const int ac = (tid & 1) * 32;       // 0 or 32 (k-chunk)
    const float*    xrow = x    + (size_t)(bm + ar) * D_SZ;
    const float*    drow = dist + (size_t)(bm + ar) * L_SZ;
    const _Float16* trow = Tt   + (size_t)(bn + ar) * K_SZ;

    // fused staging: global load -> fp32 scale -> f16 cvt -> LDS store.
    // Registers are only transiently live (no pipelining across the WMMAs).
    auto stage = [&](int s, int buf) {
        const int kc = (s & 7) * BK;     // column into x (l = s>>3)
        const float sc = drow[s >> 3];
        #pragma unroll
        for (int j = 0; j < 4; j++) {
            float4 f0 = *(const float4*)(xrow + kc + ac + 8 * j);
            float4 f1 = *(const float4*)(xrow + kc + ac + 8 * j + 4);
            v8h h;
            h[0] = (_Float16)(f0.x * sc); h[1] = (_Float16)(f0.y * sc);
            h[2] = (_Float16)(f0.z * sc); h[3] = (_Float16)(f0.w * sc);
            h[4] = (_Float16)(f1.x * sc); h[5] = (_Float16)(f1.y * sc);
            h[6] = (_Float16)(f1.z * sc); h[7] = (_Float16)(f1.w * sc);
            *(v8h*)&As[buf][ar][ac + 8 * j] = h;
        }
        const uint4* tp = (const uint4*)(trow + (size_t)s * BK + ac);
        #pragma unroll
        for (int j = 0; j < 4; j++) {
            uint4 t = tp[j];
            *(uint4*)&Bs[buf][ar][ac + 8 * j] = t;
        }
    };

    stage(0, 0);

    for (int s = 0; s < NSTAGE; s++) {
        __syncthreads();                        // buf[s&1] ready for everyone
        if (s + 1 < NSTAGE) stage(s + 1, (s + 1) & 1);  // fill other buffer
        const int buf = s & 1;

        #pragma unroll
        for (int ks = 0; ks < BK; ks += 32) {
            v8h a0l = *(const v8h*)&As[buf][wm + fm][ks + akb];
            v8h a0h = *(const v8h*)&As[buf][wm + fm][ks + akb + 16];
            v8h a1l = *(const v8h*)&As[buf][wm + 16 + fm][ks + akb];
            v8h a1h = *(const v8h*)&As[buf][wm + 16 + fm][ks + akb + 16];
            v8h b0l = *(const v8h*)&Bs[buf][wn + fm][ks + bkb];
            v8h b0h = *(const v8h*)&Bs[buf][wn + fm][ks + bkb + 16];
            v8h b1l = *(const v8h*)&Bs[buf][wn + 16 + fm][ks + bkb];
            v8h b1h = *(const v8h*)&Bs[buf][wn + 16 + fm][ks + bkb + 16];
            v16h a0 = __builtin_shufflevector(a0l, a0h, 0,1,2,3,4,5,6,7,8,9,10,11,12,13,14,15);
            v16h a1 = __builtin_shufflevector(a1l, a1h, 0,1,2,3,4,5,6,7,8,9,10,11,12,13,14,15);
            v16h b0 = __builtin_shufflevector(b0l, b0h, 0,1,2,3,4,5,6,7,8,9,10,11,12,13,14,15);
            v16h b1 = __builtin_shufflevector(b1l, b1h, 0,1,2,3,4,5,6,7,8,9,10,11,12,13,14,15);
            c00 = __builtin_amdgcn_wmma_f32_16x16x32_f16(false, a0, false, b0, (short)0, c00, false, false);
            c01 = __builtin_amdgcn_wmma_f32_16x16x32_f16(false, a0, false, b1, (short)0, c01, false, false);
            c10 = __builtin_amdgcn_wmma_f32_16x16x32_f16(false, a1, false, b0, (short)0, c10, false, false);
            c11 = __builtin_amdgcn_wmma_f32_16x16x32_f16(false, a1, false, b1, (short)0, c11, false, false);
        }
    }

    // epilogue: C VGPR r holds M=r (lanes 0-15) / M=r+8 (lanes 16-31), N=lane&15
    const int mbase = bm + wm + hi * 8;
    const int nbase = bn + wn + fm;
    #pragma unroll
    for (int r = 0; r < 8; r++) {
        out[(size_t)(mbase + r)      * W_SZ + nbase]      = c00[r];
        out[(size_t)(mbase + r)      * W_SZ + nbase + 16] = c01[r];
        out[(size_t)(mbase + 16 + r) * W_SZ + nbase]      = c10[r];
        out[(size_t)(mbase + 16 + r) * W_SZ + nbase + 16] = c11[r];
    }
}

// ---------------------------------------------------------------------------
extern "C" void kernel_launch(void* const* d_in, const int* in_sizes, int n_in,
                              void* d_out, int out_size, void* d_ws, size_t ws_size,
                              hipStream_t stream)
{
    const float* x    = (const float*)d_in[0];   // [2048, 512]
    const float* ray  = (const float*)d_in[1];   // [1, 512]
    const float* w_i  = (const float*)d_in[2];   // [1, 63]
    const float* b_i  = (const float*)d_in[3];   // [1, 63]
    const float* a_i  = (const float*)d_in[4];   // [1, 63]
    const float* T    = (const float*)d_in[5];   // [64, 512, 512] == [32768, 512]
    const int*   idx  = (const int*)d_in[6];     // [64, 6]
    float*       out  = (float*)d_out;           // [2048, 512]

    // workspace layout: Tt (f16, 32 MB) | dist (f32, 512 KB)
    _Float16* Tt   = (_Float16*)d_ws;
    float*    dist = (float*)((char*)d_ws + (size_t)W_SZ * K_SZ * sizeof(_Float16));

    prl_dist_kernel<<<B_SZ / 8, 256, 0, stream>>>(x, ray, w_i, b_i, a_i, idx, dist);
    prl_cvt_kernel<<<dim3(K_SZ / 64, W_SZ / 64), 256, 0, stream>>>(T, Tt);
    prl_gemm_f16_kernel<<<dim3(B_SZ / BM, W_SZ / BN), 128, 0, stream>>>(x, dist, Tt, out);
}
